// CombinedOrdinalLoss_83348135346708
// MI455X (gfx1250) — 8-bit matrix operand order repaired and checked by reading, compiled, not run on hardware
//
#include <hip/hip_runtime.h>

typedef __attribute__((ext_vector_type(8))) int   v8i;
typedef __attribute__((ext_vector_type(8))) float v8f;

#define LOG2E 1.4426950408889634f
#define LN2   0.6931471805599453f

__device__ __forceinline__ int match4(int w, int msplat) {
    int y = w ^ msplat;
    int a = y & 0x7f7f7f7f;
    int b = a + 0x7f7f7f7f;
    int c = (b | y) | 0x7f7f7f7f;
    return (int)(((unsigned)~c) >> 7);
}

__global__ void ordqwk_init(float* ws) {
    ws[threadIdx.x] = 0.0f;
}

__global__ __launch_bounds__(256) void ordqwk_main(
    const float* __restrict__ logits,
    const int*   __restrict__ targets,
    float* __restrict__ ws,
    int N)
{
    __shared__ float lds_cm[256];
    __shared__ float blk_s1;

    const int t0 = threadIdx.x;
    lds_cm[t0] = 0.0f;
    if (t0 == 0) blk_s1 = 0.0f;
    __syncthreads();

    const int lane   = t0 & 31;
    const int half   = lane >> 4;
    const int m      = lane & 15;
    const int msplat = m * 0x01010101;
    const int wid    = (blockIdx.x * blockDim.x + t0) >> 5;
    const int nwav   = (gridDim.x * blockDim.x) >> 5;

    v8i acc0 = {0,0,0,0,0,0,0,0};
    v8i acc1 = {0,0,0,0,0,0,0,0};
    float s1 = 0.0f;

    for (int base = wid * 128; base < N; base += nwav * 128) {
        int tcode = 0, pcode = 0;

        int   tA[4];
        float xtA[4];
#pragma unroll
        for (int j = 0; j < 4; ++j) {
            int idx  = base + j * 32 + lane;
            int idxc = (idx < N) ? idx : 0;
            int t    = targets[idxc];
            tA[j]    = t;
            int tc   = ((unsigned)t < 16u) ? t : 0;
            xtA[j]   = logits[(size_t)idxc * 16 + tc];
        }

#pragma unroll
        for (int j = 0; j < 4; ++j) {
            int idx   = base + j * 32 + lane;
            bool valid = idx < N;
            int idxc  = valid ? idx : 0;

            const float4* p = reinterpret_cast<const float4*>(logits + (size_t)idxc * 16);
            float4 q0 = p[0], q1 = p[1], q2 = p[2], q3 = p[3];
            float x[16] = {q0.x,q0.y,q0.z,q0.w, q1.x,q1.y,q1.z,q1.w,
                           q2.x,q2.y,q2.z,q2.w, q3.x,q3.y,q3.z,q3.w};

            float maxv = x[0]; int pred = 0;
#pragma unroll
            for (int k = 1; k < 16; ++k) {
                bool gt = x[k] > maxv;
                maxv = gt ? x[k] : maxv;
                pred = gt ? k   : pred;
            }

            float S = 0.0f, D = 0.0f;
#pragma unroll
            for (int k = 0; k < 16; ++k) {
                float e = __builtin_amdgcn_exp2f((x[k] - maxv) * LOG2E);
                S += e;
                D += e * (float)k;
            }

            int   t  = tA[j];
            float xt = xtA[j];

            float lnS  = __builtin_amdgcn_logf(S) * LN2;
            float ce   = lnS - (xt - maxv);
            float pcat = D * __builtin_amdgcn_rcpf(S);
            float pen  = fabsf(pcat - (float)t);
            s1 += valid ? (ce + 0.3f * pen) : 0.0f;

            int tb = (valid && t > 0 && t < 16) ? t : 0xFF;
            tcode |= tb   << (8 * j);
            pcode |= pred << (8 * j);
        }

        {
            v8i A, B;
#pragma unroll
            for (int v = 0; v < 8; ++v) {
                const int qa = ((v >> 1) << 2) | (v & 1);
                int ta0 = __builtin_amdgcn_readlane(tcode, qa);
                int ta1 = __builtin_amdgcn_readlane(tcode, qa | 2);
                A[v] = match4(half ? ta1 : ta0, msplat);
                const int qb = ((v >> 1) << 2) | (v & 1);
                int pb0 = __builtin_amdgcn_readlane(pcode, qb);
                int pb1 = __builtin_amdgcn_readlane(pcode, qb | 2);
                B[v] = match4(half ? pb1 : pb0, msplat);
            }
            acc0 = __builtin_amdgcn_wmma_i32_16x16x64_iu8(false, A, false, B, acc0, false, false);
        }
        {
            v8i A, B;
#pragma unroll
            for (int v = 0; v < 8; ++v) {
                const int qa = (((v >> 1) << 2) | (v & 1)) | 16;
                int ta0 = __builtin_amdgcn_readlane(tcode, qa);
                int ta1 = __builtin_amdgcn_readlane(tcode, qa | 2);
                A[v] = match4(half ? ta1 : ta0, msplat);
                const int qb = (((v >> 1) << 2) | (v & 1)) | 16;
                int pb0 = __builtin_amdgcn_readlane(pcode, qb);
                int pb1 = __builtin_amdgcn_readlane(pcode, qb | 2);
                B[v] = match4(half ? pb1 : pb0, msplat);
            }
            acc1 = __builtin_amdgcn_wmma_i32_16x16x64_iu8(false, A, false, B, acc1, false, false);
        }
    }

    v8f cms;
#pragma unroll
    for (int r = 0; r < 8; ++r) cms[r] = (float)(acc0[r] + acc1[r]);

#pragma unroll
    for (int d = 16; d > 0; d >>= 1) s1 += __shfl_xor(s1, d, 32);

#pragma unroll
    for (int r = 0; r < 8; ++r)
        atomicAdd(&lds_cm[(r + 8 * half) * 16 + m], cms[r]);
    if (lane == 0) atomicAdd(&blk_s1, s1);
    __syncthreads();

    atomicAdd(&ws[t0], lds_cm[t0]);
    if (t0 == 0) ws[256 + blockIdx.x] = blk_s1;
}

__global__ __launch_bounds__(256) void ordqwk_final(
    const float* __restrict__ ws, float* __restrict__ out, int nblocks, float invN)
{
    __shared__ float c[256];
    __shared__ float red[256];
    __shared__ float rs[16];
    __shared__ float csum[16];
    __shared__ float scal[3];

    int t = threadIdx.x;
    float cv = ws[t];
    c[t] = cv;

    float s1p = 0.0f;
    for (int k = t; k < nblocks; k += 256) s1p += ws[256 + k];

    red[t] = s1p; __syncthreads();
    for (int s = 128; s > 0; s >>= 1) { if (t < s) red[t] += red[t + s]; __syncthreads(); }
    if (t == 0) scal[0] = red[0];
    __syncthreads();

    red[t] = cv; __syncthreads();
    for (int s = 128; s > 0; s >>= 1) { if (t < s) red[t] += red[t + s]; __syncthreads(); }
    if (t == 0) scal[1] = red[0];
    __syncthreads();

    if (t < 16)      { float r = 0.f; for (int j = 0; j < 16; ++j) r += c[t * 16 + j];       rs[t] = r; }
    else if (t < 32) { int j = t - 16; float r = 0.f; for (int i2 = 0; i2 < 16; ++i2) r += c[i2 * 16 + j]; csum[j] = r; }
    __syncthreads();

    int i = t >> 4, j = t & 15;
    float d = (float)(i - j);
    float w = 1.0f - d * d * (1.0f / 225.0f);

    red[t] = w * cv; __syncthreads();
    for (int s = 128; s > 0; s >>= 1) { if (t < s) red[t] += red[t + s]; __syncthreads(); }
    if (t == 0) scal[2] = red[0];
    __syncthreads();

    red[t] = w * rs[i] * csum[j]; __syncthreads();
    for (int s = 128; s > 0; s >>= 1) { if (t < s) red[t] += red[t + s]; __syncthreads(); }

    if (t == 0) {
        float s1  = scal[0];
        float n   = scal[1];
        float num = scal[2];
        float den = red[0];
        float ce_loss = s1 * invN;
        float qwk      = (den == 0.0f) ? 0.0f : (num * n / den);
        float qwk_loss = (n == 0.0f) ? 0.0f : (1.0f - qwk);
        out[0] = 1.0f * ce_loss + 0.5f * qwk_loss;
        out[1] = ce_loss;
        out[2] = qwk_loss;
    }
}

extern "C" void kernel_launch(void* const* d_in, const int* in_sizes, int n_in,
                              void* d_out, int out_size, void* d_ws, size_t ws_size,
                              hipStream_t stream) {
    const float* logits  = (const float*)d_in[0];
    const int*   targets = (const int*)d_in[1];
    float* ws  = (float*)d_ws;
    float* out = (float*)d_out;
    const int N = in_sizes[1];
    const int blocks = 1024, threads = 256;

    ordqwk_init<<<1, 256, 0, stream>>>(ws);
    ordqwk_main<<<blocks, threads, 0, stream>>>(logits, targets, ws, N);
    ordqwk_final<<<1, 256, 0, stream>>>(ws, out, blocks, 1.0f / (float)N);
}
